// MoEFusion_19112604467910
// MI455X (gfx1250) — matrix operand order repaired and checked by reading, not compiled, not run on hardware
//
#include <hip/hip_runtime.h>

typedef float  v8f   __attribute__((ext_vector_type(8)));
typedef __bf16 v8bf  __attribute__((ext_vector_type(8)));
typedef __bf16 v16bf __attribute__((ext_vector_type(16)));

#define B_    2
#define C_    32
#define MC_   128
#define O_    128
#define E_    4
#define S_    32768
#define NTILE 64
#define KSTR  136    // LDS column stride in bf16 elements (128 + 8 pad)

// ---------------------------------------------------------------------------
// Kernel 1: global average pool over spatial dim for each (b, c) channel.
// ---------------------------------------------------------------------------
__global__ __launch_bounds__(256) void pool_kernel(
    const float* __restrict__ f0, const float* __restrict__ f1,
    const float* __restrict__ f2, const float* __restrict__ f3,
    float* __restrict__ pooled)
{
    __shared__ float red[256];
    int bc = blockIdx.x;          // 0..255
    int b  = bc >> 7;
    int c  = bc & 127;
    const float* src = (c < 32) ? f0 : (c < 64) ? f1 : (c < 96) ? f2 : f3;
    src += ((size_t)(b * C_ + (c & 31))) * S_;

    float s = 0.f;
    for (int i = threadIdx.x; i < S_; i += 256) s += src[i];
    red[threadIdx.x] = s;
    __syncthreads();
    for (int off = 128; off > 0; off >>= 1) {
        if ((int)threadIdx.x < off) red[threadIdx.x] += red[threadIdx.x + off];
        __syncthreads();
    }
    if (threadIdx.x == 0) pooled[bc] = red[0] * (1.0f / (float)S_);
}

// ---------------------------------------------------------------------------
// Kernel 2: router. 1 wave. logits -> softmax -> top-2 -> weights + aux loss.
// ---------------------------------------------------------------------------
__global__ __launch_bounds__(32) void router_kernel(
    const float* __restrict__ pooled, const float* __restrict__ Wr,
    const float* __restrict__ br,
    int* __restrict__ selIdx, float* __restrict__ selW,
    float* __restrict__ auxOut)
{
    __shared__ float logits[B_ * E_];
    int lane = threadIdx.x;

    for (int p = 0; p < B_ * E_; ++p) {
        int b = p / E_, e = p % E_;
        float s = 0.f;
        for (int c = lane; c < MC_; c += 32)
            s += pooled[b * MC_ + c] * Wr[e * MC_ + c];
        for (int off = 16; off > 0; off >>= 1)
            s += __shfl_xor(s, off, 32);
        if (lane == 0) logits[p] = s + br[e];
    }
    __syncthreads();

    if (lane == 0) {
        float probs[B_][E_];
        float p_e[E_] = {0.f, 0.f, 0.f, 0.f};
        float f_e[E_] = {0.f, 0.f, 0.f, 0.f};
        for (int b = 0; b < B_; ++b) {
            float mx = logits[b * E_];
            for (int e = 1; e < E_; ++e) mx = fmaxf(mx, logits[b * E_ + e]);
            float sum = 0.f;
            for (int e = 0; e < E_; ++e) {
                probs[b][e] = expf(logits[b * E_ + e] - mx);
                sum += probs[b][e];
            }
            for (int e = 0; e < E_; ++e) {
                probs[b][e] /= sum;
                p_e[e] += probs[b][e] * (1.0f / (float)B_);
            }
            int i0 = 0;
            for (int e = 1; e < E_; ++e) if (probs[b][e] > probs[b][i0]) i0 = e;
            int i1 = (i0 == 0) ? 1 : 0;
            for (int e = 0; e < E_; ++e)
                if (e != i0 && probs[b][e] > probs[b][i1]) i1 = e;
            float w0 = probs[b][i0], w1 = probs[b][i1];
            float inv = 1.0f / (w0 + w1);
            selIdx[b * 2 + 0] = i0;  selIdx[b * 2 + 1] = i1;
            selW[b * 2 + 0] = w0 * inv;  selW[b * 2 + 1] = w1 * inv;
            f_e[i0] += 1.0f / (float)B_;
        }
        float aux = 0.f;
        for (int e = 0; e < E_; ++e) aux += f_e[e] * p_e[e];
        *auxOut = aux * (float)E_;
    }
}

// ---------------------------------------------------------------------------
// Kernel 3: pre-swizzle Wc into bf16 hi/lo planes, laid out exactly as the
// 16-bit A-matrix 16x32 WMMA fragment wants per lane:
//   fragment id = (((e*4 + kb)*128 + m)*2 + half), 16 bf16 per fragment,
//   element i -> K = kb*32 + (p<4 ? 2p+w+8*half : 16+2(p-4)+w+8*half)
// ---------------------------------------------------------------------------
__global__ __launch_bounds__(256) void prep_wc_kernel(
    const float* __restrict__ Wc, __bf16* __restrict__ WcHi, __bf16* __restrict__ WcLo)
{
    int id   = blockIdx.x * 256 + threadIdx.x;   // 0 .. 65535
    int i    = id & 15;
    int half = (id >> 4) & 1;
    int m    = (id >> 5) & 127;
    int kb   = (id >> 12) & 3;
    int e    = (id >> 14) & 3;
    int p = i >> 1, w = i & 1;
    int koff = (p < 4) ? (2 * p + w + 8 * half) : (16 + 2 * (p - 4) + w + 8 * half);
    float v = Wc[((size_t)(e * MC_ + m)) * MC_ + kb * 32 + koff];
    __bf16 h = (__bf16)v;
    __bf16 l = (__bf16)(v - (float)h);
    WcHi[id] = h;
    WcLo[id] = l;
}

// ---------------------------------------------------------------------------
// bf16 fragment loaders (two 16B chunks -> one v16bf)
// ---------------------------------------------------------------------------
__device__ __forceinline__ v16bf ldfrag(const __bf16* p)
{
    v8bf lo = *(const v8bf*)p;
    v8bf hi = *(const v8bf*)(p + 8);
    return __builtin_shufflevector(lo, hi, 0, 1, 2, 3, 4, 5, 6, 7,
                                           8, 9, 10, 11, 12, 13, 14, 15);
}

__device__ __forceinline__ v8f wmma_bf16(v16bf a, v16bf b, v8f c)
{
    return __builtin_amdgcn_wmma_f32_16x16x32_bf16(
        false, a, false, b, (short)0, c, false, false);
}

// ---------------------------------------------------------------------------
// Kernel 4: fused dual-expert GEMM (bf16x3 split, fp32 accumulate)
//           + BN + ReLU + weighted top-2 combine.
// grid = B * (S/NTILE) blocks of 256 threads (8 waves).
// ---------------------------------------------------------------------------
__global__ __launch_bounds__(256) void moe_gemm_kernel(
    const float* __restrict__ f0, const float* __restrict__ f1,
    const float* __restrict__ f2, const float* __restrict__ f3,
    const __bf16* __restrict__ WcHi, const __bf16* __restrict__ WcLo,
    const float* __restrict__ gamma, const float* __restrict__ beta,
    const float* __restrict__ rmean, const float* __restrict__ rvar,
    const int* __restrict__ selIdx, const float* __restrict__ selW,
    float* __restrict__ out)
{
    // column-major bf16 x tile: [n][k], stride KSTR to spread banks
    __shared__ __bf16 xhi[NTILE * KSTR];   // 17408 B
    __shared__ __bf16 xlo[NTILE * KSTR];   // 17408 B

    int blk = blockIdx.x;
    int b   = blk / (S_ / NTILE);
    int s0  = (blk % (S_ / NTILE)) * NTILE;

    // ---- stage x tile: fp32 global -> bf16 hi/lo split -> LDS (col-major) ----
    for (int i = threadIdx.x; i < MC_ * NTILE; i += 256) {
        int c = i >> 6;          // k (channel), 0..127
        int j = i & 63;          // n (spatial), 0..63  -> coalesced reads
        const float* src = (c < 32) ? f0 : (c < 64) ? f1 : (c < 96) ? f2 : f3;
        float v = src[((size_t)(b * C_ + (c & 31))) * S_ + s0 + j];
        __bf16 h = (__bf16)v;
        __bf16 l = (__bf16)(v - (float)h);
        xhi[j * KSTR + c] = h;
        xlo[j * KSTR + c] = l;
    }
    __syncthreads();

    int wave   = threadIdx.x >> 5;   // 0..7 -> M tile
    int lane   = threadIdx.x & 31;
    int lm     = lane & 15;
    int half   = lane >> 4;
    int m_base = wave * 16;

    int   e0 = selIdx[b * 2 + 0];
    int   e1 = selIdx[b * 2 + 1];
    float w0 = selW[b * 2 + 0];
    float w1 = selW[b * 2 + 1];

    v8f acc[2][4] = {};

    #pragma unroll
    for (int kb = 0; kb < 4; ++kb) {
        size_t a0off = ((((size_t)(e0 * 4 + kb)) * MC_ + (m_base + lm)) * 2 + half) * 16;
        size_t a1off = ((((size_t)(e1 * 4 + kb)) * MC_ + (m_base + lm)) * 2 + half) * 16;
        v16bf a0h = ldfrag(WcHi + a0off);
        v16bf a0l = ldfrag(WcLo + a0off);
        v16bf a1h = ldfrag(WcHi + a1off);
        v16bf a1l = ldfrag(WcLo + a1off);

        int bbase = kb * 32 + 8 * half;
        #pragma unroll
        for (int nt = 0; nt < 4; ++nt) {
            int n = nt * 16 + lm;
            v16bf bh = __builtin_shufflevector(*(const v8bf*)&xhi[n * KSTR + bbase], *(const v8bf*)&xhi[n * KSTR + bbase + 16], 0, 1, 2, 3, 4, 5, 6, 7, 8, 9, 10, 11, 12, 13, 14, 15);
            v16bf bl = __builtin_shufflevector(*(const v8bf*)&xlo[n * KSTR + bbase], *(const v8bf*)&xlo[n * KSTR + bbase + 16], 0, 1, 2, 3, 4, 5, 6, 7, 8, 9, 10, 11, 12, 13, 14, 15);
            // bf16x3: hi*hi + hi*lo + lo*hi  (fp32 accumulate)
            acc[0][nt] = wmma_bf16(a0h, bh, acc[0][nt]);
            acc[0][nt] = wmma_bf16(a0h, bl, acc[0][nt]);
            acc[0][nt] = wmma_bf16(a0l, bh, acc[0][nt]);
            acc[1][nt] = wmma_bf16(a1h, bh, acc[1][nt]);
            acc[1][nt] = wmma_bf16(a1h, bl, acc[1][nt]);
            acc[1][nt] = wmma_bf16(a1l, bh, acc[1][nt]);
        }
    }

    // ---- BN (eval) scale/shift per (expert, out-channel) ----
    float inv0[8], bia0[8], inv1[8], bia1[8];
    #pragma unroll
    for (int r = 0; r < 8; ++r) {
        int o = m_base + r + 8 * half;
        float iv = gamma[e0 * O_ + o] * rsqrtf(rvar[e0 * O_ + o] + 1e-5f);
        inv0[r] = iv;
        bia0[r] = beta[e0 * O_ + o] - rmean[e0 * O_ + o] * iv;
        iv = gamma[e1 * O_ + o] * rsqrtf(rvar[e1 * O_ + o] + 1e-5f);
        inv1[r] = iv;
        bia1[r] = beta[e1 * O_ + o] - rmean[e1 * O_ + o] * iv;
    }

    // ---- ReLU + top-2 weighted combine, coalesced store ----
    #pragma unroll
    for (int nt = 0; nt < 4; ++nt) {
        #pragma unroll
        for (int r = 0; r < 8; ++r) {
            int o = m_base + r + 8 * half;
            float v0 = fmaxf(acc[0][nt][r] * inv0[r] + bia0[r], 0.f);
            float v1 = fmaxf(acc[1][nt][r] * inv1[r] + bia1[r], 0.f);
            out[((size_t)(b * O_ + o)) * S_ + s0 + nt * 16 + lm] = w0 * v0 + w1 * v1;
        }
    }
}

// ---------------------------------------------------------------------------
extern "C" void kernel_launch(void* const* d_in, const int* in_sizes, int n_in,
                              void* d_out, int out_size, void* d_ws, size_t ws_size,
                              hipStream_t stream)
{
    const float* f0    = (const float*)d_in[0];
    const float* f1    = (const float*)d_in[1];
    const float* f2    = (const float*)d_in[2];
    const float* f3    = (const float*)d_in[3];
    const float* Wc    = (const float*)d_in[4];
    const float* gamma = (const float*)d_in[5];
    const float* beta  = (const float*)d_in[6];
    const float* rmean = (const float*)d_in[7];
    const float* rvar  = (const float*)d_in[8];
    const float* Wr    = (const float*)d_in[9];
    const float* br    = (const float*)d_in[10];
    // d_in[11] = top_k (fixed at 2 in the reference harness)

    float*  pooled = (float*)d_ws;                                  // 1 KB
    int*    selIdx = (int*)((char*)d_ws + 1024);
    float*  selW   = (float*)((char*)d_ws + 1024 + 64);
    __bf16* WcHi   = (__bf16*)((char*)d_ws + 2048);                 // 128 KB
    __bf16* WcLo   = (__bf16*)((char*)d_ws + 2048 + 131072);        // 128 KB
    float*  outp   = (float*)d_out;
    float*  aux    = outp + (size_t)B_ * O_ * S_;                   // aux_loss slot

    pool_kernel<<<B_ * MC_, 256, 0, stream>>>(f0, f1, f2, f3, pooled);
    router_kernel<<<1, 32, 0, stream>>>(pooled, Wr, br, selIdx, selW, aux);
    prep_wc_kernel<<<(E_ * O_ * MC_) / 256, 256, 0, stream>>>(Wc, WcHi, WcLo);
    moe_gemm_kernel<<<B_ * (S_ / NTILE), 256, 0, stream>>>(
        f0, f1, f2, f3, WcHi, WcLo, gamma, beta, rmean, rvar, selIdx, selW, outp);
}
